// D_net_5136780886318
// MI455X (gfx1250) — matrix operand order repaired and checked by reading, not compiled, not run on hardware
//
#include <hip/hip_runtime.h>
#include <hip/hip_bf16.h>

typedef __attribute__((ext_vector_type(16))) __bf16 v16bf;
typedef __attribute__((ext_vector_type(8)))  float  v8f;

#define BN_EPS 1e-5f
#define KMAX 64

// ---------------------------------------------------------------------------
// (B, N, C) -> (B, C, N)   (channel-major f32, for the kNN kernel)
// ---------------------------------------------------------------------------
__global__ void k_transpose(const float* __restrict__ x, float* __restrict__ xt,
                            int B, int N, int C) {
  int i = blockIdx.x * blockDim.x + threadIdx.x;
  int total = B * N * C;
  if (i >= total) return;
  int c = i % C;
  int n = (i / C) % N;
  int b = i / (C * N);
  xt[((size_t)b * C + c) * N + n] = x[i];
}

// ---------------------------------------------------------------------------
// (B, C, N) f32 -> (B, N, CP) bf16, CP = C padded to multiple of 32, zeros in
// the pad. Point-major so 16 consecutive channels are one 32-byte load.
// ---------------------------------------------------------------------------
__global__ void k_tobf16(const float* __restrict__ h, __bf16* __restrict__ xT,
                         int B, int N, int C, int CP) {
  int i = blockIdx.x * blockDim.x + threadIdx.x;
  if (i >= B * N * CP) return;
  int c = i % CP;
  int n = (i / CP) % N;
  int b = i / (CP * N);
  float v = (c < C) ? h[((size_t)b * C + c) * N + n] : 0.f;
  xT[i] = (__bf16)v;
}

// ---------------------------------------------------------------------------
// kNN (top-k of -squared-distance) per (b, n). One 256-thread block per row.
// dist(m) = 2*dot(x_n, x_m) - |x_m|^2   (-|x_n|^2 is constant in m)
// Selection: K rounds of wave32 shuffle-reduce + 8-leader LDS merge.
// ---------------------------------------------------------------------------
__global__ void __launch_bounds__(256)
k_topk(const float* __restrict__ xt, int* __restrict__ idx,
       const int* __restrict__ kptr, int N, int C) {
  const int K = *kptr;
  const int b = blockIdx.y;
  const int n = blockIdx.x;
  const int t = threadIdx.x;
  const int lane = t & 31;
  const int wave = t >> 5;

  __shared__ float dist[1024];
  __shared__ float ctr[128];
  __shared__ float rv[8];
  __shared__ int   ri[8];

  const float* xb = xt + (size_t)b * C * N;

  for (int c = t; c < C; c += 256) ctr[c] = xb[(size_t)c * N + n];
  __syncthreads();

  for (int m = t; m < N; m += 256) {
    float dot = 0.f, sq = 0.f;
    for (int c = 0; c < C; ++c) {
      float xm = xb[(size_t)c * N + m];
      dot += xm * ctr[c];
      sq  += xm * xm;
    }
    dist[m] = 2.f * dot - sq;
  }
  __syncthreads();

  for (int r = 0; r < K; ++r) {
    float best = -__builtin_inff();
    int   bi   = 0x7fffffff;
    for (int m = t; m < N; m += 256) {
      float d = dist[m];
      if (d > best || (d == best && m < bi)) { best = d; bi = m; }
    }
    // wave32 reduction, no barriers
    for (int s = 16; s > 0; s >>= 1) {
      float ov = __shfl_xor(best, s, 32);
      int   oi = __shfl_xor(bi,   s, 32);
      if (ov > best || (ov == best && oi < bi)) { best = ov; bi = oi; }
    }
    if (lane == 0) { rv[wave] = best; ri[wave] = bi; }
    __syncthreads();
    if (t == 0) {
      float fb = rv[0]; int fi = ri[0];
      for (int w = 1; w < 8; ++w)
        if (rv[w] > fb || (rv[w] == fb && ri[w] < fi)) { fb = rv[w]; fi = ri[w]; }
      idx[((size_t)b * N + n) * KMAX + r] = fi;
      dist[fi] = -__builtin_inff();
    }
    __syncthreads();
  }
}

// ---------------------------------------------------------------------------
// Fused EdgeConv via the identity
//   h(o,n,k) = Wd . x_j  +  (Wc - Wd) . x_n
// where W = [Wd | Wc].  u(o,n) = (Wc-Wd).x_n is computed once per tile and
// used as the WMMA C operand for every neighbor; per-k work is Wd.x_j only.
// One wave32 per 16(O) x 16(N) tile; v_wmma_f32_16x16x32_bf16, f32 acc.
// Then BN + LeakyReLU(0.2) + max over k.
// ---------------------------------------------------------------------------
template <int CIN>
__global__ void __launch_bounds__(32)
k_edgeconv(const __bf16* __restrict__ xT,  // (B, N, CP) bf16, zero-padded
           const int*    __restrict__ idx, // (B, N, KMAX)
           const int*    __restrict__ kptr,
           const float*  __restrict__ W,   // (O, 2*CIN) = [Wd | Wc]
           const float* __restrict__ gg, const float* __restrict__ bb,
           const float* __restrict__ mm, const float* __restrict__ vv,
           float* __restrict__ out,        // (B, O, N)
           int N, int O) {
  constexpr int CP  = ((CIN + 31) / 32) * 32;  // padded channel count
  constexpr int NCH = CP / 32;                 // 32-wide K chunks
  constexpr int CI  = 2 * CIN;

  const int K      = *kptr;
  const int lane   = threadIdx.x;
  const int half   = lane >> 4;     // 0: lanes 0-15, 1: lanes 16-31
  const int l      = lane & 15;
  const int n_base = blockIdx.x * 16;
  const int o_base = blockIdx.y * 16;
  const int b      = blockIdx.z;

  const int n = n_base + l;
  const __bf16* xTb = xT + (size_t)b * N * CP;

  // ---- A fragments. Contiguous K layout (matches B): lanes 0-15 hold row M=l,
  // K = {c0..c0+15}; lanes 16-31 hold K = {c0+16..c0+31}.
  v16bf aD[NCH];    // Wd
  v16bf aU[NCH];    // Wc - Wd
  const int arow = o_base + l;
  const float* Wr = W + (size_t)arow * CI;
#pragma unroll
  for (int ch = 0; ch < NCH; ++ch) {
    const int c0 = ch * 32;
    v16bf ad, au;
#pragma unroll
    for (int j = 0; j < 8; ++j) {
      int c1 = c0 + half * 16 + j;
      int c2 = c0 + half * 16 + 8 + j;
      float wd1 = (c1 < CIN) ? Wr[c1] : 0.f;
      float wc1 = (c1 < CIN) ? Wr[CIN + c1] : 0.f;
      float wd2 = (c2 < CIN) ? Wr[c2] : 0.f;
      float wc2 = (c2 < CIN) ? Wr[CIN + c2] : 0.f;
      ad[j] = (__bf16)wd1;      ad[8 + j] = (__bf16)wd2;
      au[j] = (__bf16)(wc1 - wd1); au[8 + j] = (__bf16)(wc2 - wd2);
    }
    aD[ch] = ad;
    aU[ch] = au;
  }

  // ---- u(o, n) = (Wc - Wd) . x_n  (k-invariant seed accumulator)
  v8f acc0 = (v8f)0.f;
#pragma unroll
  for (int ch = 0; ch < NCH; ++ch) {
    // B layout (32x16 bf16): lanes 0-15 -> K rows c0..c0+15 of column n,
    // lanes 16-31 -> K rows c0+16..c0+31. Contiguous bf16 -> 32B vector load.
    const int c0 = ch * 32 + half * 16;
    v16bf bfrag = *(const v16bf*)(xTb + (size_t)n * CP + c0);
    acc0 = __builtin_amdgcn_wmma_f32_16x16x32_bf16(
        false, aU[ch], false, bfrag, (short)0, acc0, false, false);
  }

  // ---- BN params. C/D layout: VGPR r -> M = r + 8*half, N = l.
  float sc[8], sh[8];
#pragma unroll
  for (int r = 0; r < 8; ++r) {
    int oc = o_base + half * 8 + r;
    float s = gg[oc] * __frsqrt_rn(vv[oc] + BN_EPS);
    sc[r] = s;
    sh[r] = bb[oc] - mm[oc] * s;
  }

  float run[8];
#pragma unroll
  for (int r = 0; r < 8; ++r) run[r] = -__builtin_inff();

  const int* idxrow = idx + ((size_t)b * N + n) * KMAX;
  for (int kk = 0; kk < K; ++kk) {
    const int j = idxrow[kk];
    v8f acc = acc0;                       // seeded with (Wc-Wd).x_n
#pragma unroll
    for (int ch = 0; ch < NCH; ++ch) {
      const int c0 = ch * 32 + half * 16;
      v16bf bfrag = *(const v16bf*)(xTb + (size_t)j * CP + c0);
      acc = __builtin_amdgcn_wmma_f32_16x16x32_bf16(
          false, aD[ch], false, bfrag, (short)0, acc, false, false);
    }
#pragma unroll
    for (int r = 0; r < 8; ++r) {
      float y = acc[r] * sc[r] + sh[r];
      y = (y >= 0.f) ? y : 0.2f * y;
      run[r] = fmaxf(run[r], y);
    }
  }

#pragma unroll
  for (int r = 0; r < 8; ++r) {
    int oc = o_base + half * 8 + r;
    out[((size_t)b * O + oc) * N + n] = run[r];
  }
}

// ---------------------------------------------------------------------------
// Global max pool over N: f[b, 0:128] = max_n h3, f[b, 128:384] = max_n h4
// ---------------------------------------------------------------------------
__global__ void k_pool(const float* __restrict__ h3, const float* __restrict__ h4,
                       float* __restrict__ f, int B, int N) {
  int i = blockIdx.x * blockDim.x + threadIdx.x;
  if (i >= B * 384) return;
  int c = i % 384, b = i / 384;
  const float* src = (c < 128) ? (h3 + ((size_t)b * 128 + c) * N)
                               : (h4 + ((size_t)b * 256 + (c - 128)) * N);
  float m = -__builtin_inff();
  for (int nn = 0; nn < N; ++nn) m = fmaxf(m, src[nn]);
  f[i] = m;
}

// ---------------------------------------------------------------------------
// MLP head: (8,384) -> 128 (BN+ReLU) -> 16 (BN+ReLU) -> 1. One block.
// ---------------------------------------------------------------------------
__global__ void __launch_bounds__(256)
k_head(const float* __restrict__ f,
       const float* f1w, const float* f1b,
       const float* n1g, const float* n1b, const float* n1m, const float* n1v,
       const float* f2w, const float* f2b,
       const float* n2g, const float* n2b, const float* n2m, const float* n2v,
       const float* f3w, const float* f3b,
       float* __restrict__ outp) {
  __shared__ float s0[8 * 384];
  __shared__ float s1[8 * 128];
  __shared__ float s2[8 * 16];
  const int t = threadIdx.x;

  for (int i = t; i < 8 * 384; i += 256) s0[i] = f[i];
  __syncthreads();

  for (int i = t; i < 8 * 128; i += 256) {
    int b = i / 128, oc = i % 128;
    float acc = f1b[oc];
    for (int c = 0; c < 384; ++c) acc += s0[b * 384 + c] * f1w[oc * 384 + c];
    float s = n1g[oc] * __frsqrt_rn(n1v[oc] + BN_EPS);
    float y = (acc - n1m[oc]) * s + n1b[oc];
    s1[i] = fmaxf(y, 0.f);
  }
  __syncthreads();

  for (int i = t; i < 8 * 16; i += 256) {
    int b = i / 16, oc = i % 16;
    float acc = f2b[oc];
    for (int c = 0; c < 128; ++c) acc += s1[b * 128 + c] * f2w[oc * 128 + c];
    float s = n2g[oc] * __frsqrt_rn(n2v[oc] + BN_EPS);
    float y = (acc - n2m[oc]) * s + n2b[oc];
    s2[i] = fmaxf(y, 0.f);
  }
  __syncthreads();

  if (t < 8) {
    float acc = f3b[0];
    for (int c = 0; c < 16; ++c) acc += s2[t * 16 + c] * f3w[c];
    outp[t] = acc;
  }
}

// ---------------------------------------------------------------------------
extern "C" void kernel_launch(void* const* d_in, const int* in_sizes, int n_in,
                              void* d_out, int out_size, void* d_ws, size_t ws_size,
                              hipStream_t stream) {
  const int B = 8, N = 1024;

  const float* x   = (const float*)d_in[0];
  const float* c1w = (const float*)d_in[1];
  const float* c1g = (const float*)d_in[2];
  const float* c1b = (const float*)d_in[3];
  const float* c1m = (const float*)d_in[4];
  const float* c1v = (const float*)d_in[5];
  const float* c2w = (const float*)d_in[6];
  const float* c2g = (const float*)d_in[7];
  const float* c2b = (const float*)d_in[8];
  const float* c2m = (const float*)d_in[9];
  const float* c2v = (const float*)d_in[10];
  const float* c3w = (const float*)d_in[11];
  const float* c3g = (const float*)d_in[12];
  const float* c3b = (const float*)d_in[13];
  const float* c3m = (const float*)d_in[14];
  const float* c3v = (const float*)d_in[15];
  const float* c4w = (const float*)d_in[16];
  const float* c4g = (const float*)d_in[17];
  const float* c4b = (const float*)d_in[18];
  const float* c4m = (const float*)d_in[19];
  const float* c4v = (const float*)d_in[20];
  const float* f1w = (const float*)d_in[21];
  const float* f1b = (const float*)d_in[22];
  const float* n1g = (const float*)d_in[23];
  const float* n1b = (const float*)d_in[24];
  const float* n1m = (const float*)d_in[25];
  const float* n1v = (const float*)d_in[26];
  const float* f2w = (const float*)d_in[27];
  const float* f2b = (const float*)d_in[28];
  const float* n2g = (const float*)d_in[29];
  const float* n2b = (const float*)d_in[30];
  const float* n2m = (const float*)d_in[31];
  const float* n2v = (const float*)d_in[32];
  const float* f3w = (const float*)d_in[33];
  const float* f3b = (const float*)d_in[34];
  const int*   kp  = (const int*)d_in[35];

  // workspace layout
  float* ws = (float*)d_ws;
  size_t off = 0;
  float* xt    = ws + off; off += (size_t)B * 3   * N;
  float* h1    = ws + off; off += (size_t)B * 64  * N;
  float* h2    = ws + off; off += (size_t)B * 64  * N;
  float* h3    = ws + off; off += (size_t)B * 128 * N;
  float* h4    = ws + off; off += (size_t)B * 256 * N;
  float* fpool = ws + off; off += (size_t)B * 384;
  off = (off + 63) & ~(size_t)63;                 // align
  int*   idx   = (int*)(ws + off); off += (size_t)B * N * KMAX;
  off = (off + 63) & ~(size_t)63;
  __bf16* xTb  = (__bf16*)(ws + off);             // (B, N, 128) bf16, reused

  dim3 tkGrid(N, B);
  auto cdiv = [](int a, int b) { return (a + b - 1) / b; };

  // layer 0: transpose input to channel-major
  k_transpose<<<cdiv(B * N * 3, 256), 256, 0, stream>>>(x, xt, B, N, 3);

  // EdgeConv 1: 3 -> 64  (CP = 32)
  k_topk<<<tkGrid, 256, 0, stream>>>(xt, idx, kp, N, 3);
  k_tobf16<<<cdiv(B * N * 32, 256), 256, 0, stream>>>(xt, xTb, B, N, 3, 32);
  k_edgeconv<3><<<dim3(N / 16, 64 / 16, B), 32, 0, stream>>>(
      xTb, idx, kp, c1w, c1g, c1b, c1m, c1v, h1, N, 64);

  // EdgeConv 2: 64 -> 64  (CP = 64)
  k_topk<<<tkGrid, 256, 0, stream>>>(h1, idx, kp, N, 64);
  k_tobf16<<<cdiv(B * N * 64, 256), 256, 0, stream>>>(h1, xTb, B, N, 64, 64);
  k_edgeconv<64><<<dim3(N / 16, 64 / 16, B), 32, 0, stream>>>(
      xTb, idx, kp, c2w, c2g, c2b, c2m, c2v, h2, N, 64);

  // EdgeConv 3: 64 -> 128  (CP = 64)
  k_topk<<<tkGrid, 256, 0, stream>>>(h2, idx, kp, N, 64);
  k_tobf16<<<cdiv(B * N * 64, 256), 256, 0, stream>>>(h2, xTb, B, N, 64, 64);
  k_edgeconv<64><<<dim3(N / 16, 128 / 16, B), 32, 0, stream>>>(
      xTb, idx, kp, c3w, c3g, c3b, c3m, c3v, h3, N, 128);

  // EdgeConv 4: 128 -> 256  (CP = 128)
  k_topk<<<tkGrid, 256, 0, stream>>>(h3, idx, kp, N, 128);
  k_tobf16<<<cdiv(B * N * 128, 256), 256, 0, stream>>>(h3, xTb, B, N, 128, 128);
  k_edgeconv<128><<<dim3(N / 16, 256 / 16, B), 32, 0, stream>>>(
      xTb, idx, kp, c4w, c4g, c4b, c4m, c4v, h4, N, 256);

  // pool + head
  k_pool<<<cdiv(B * 384, 256), 256, 0, stream>>>(h3, h4, fpool, B, N);
  k_head<<<1, 256, 0, stream>>>(fpool, f1w, f1b, n1g, n1b, n1m, n1v,
                                f2w, f2b, n2g, n2b, n2m, n2v, f3w, f3b,
                                (float*)d_out);
}
